// GAT_30167850287504
// MI455X (gfx1250) — matrix operand order repaired and checked by reading, not compiled, not run on hardware
//
#include <hip/hip_runtime.h>
#include <hip/hip_bf16.h>

typedef __attribute__((ext_vector_type(16))) _Float16 v16h;
typedef __attribute__((ext_vector_type(8)))  _Float16 v8h;
typedef __attribute__((ext_vector_type(8)))  float    v8f;

#define S_   100
#define D_   30
#define T_   40
#define FT_  512
#define FP_  3
#define H_   64
#define NH_  4
#define DH_  16
#define G3   (3*H_)        // 192
#define G4   (4*H_)        // 256
#define M_TL (D_*T_)       // 1200 rows of text per stock

__device__ __forceinline__ float sigf(float x) { return 1.0f/(1.0f+__expf(-x)); }

// ---------------------------------------------------------------------------
// Kernel 0: one-shot f32 -> f16 conversion of tl_Wall (S,4H,FT).
// 13.1M elements; trivial bandwidth (~79 MB round trip) but removes 75x
// re-conversion of the B operand inside the WMMA GEMM.
// ---------------------------------------------------------------------------
__global__ __launch_bounds__(256) void cvt_wall_f16(
    const float* __restrict__ in, _Float16* __restrict__ out) {
  const size_t i = ((size_t)blockIdx.x * 256 + threadIdx.x) * 4;
#pragma unroll
  for (int e = 0; e < 4; ++e) out[i + e] = (_Float16)in[i + e];
}

__device__ __forceinline__ v16h loadA_f32(const float* __restrict__ xp) {
  // 16x32 f16 A fragment for this lane: xp[0..15] contiguous
  v16h a;
#pragma unroll
  for (int e = 0; e < 8; ++e) a[e]   = (_Float16)xp[e];
#pragma unroll
  for (int e = 0; e < 8; ++e) a[8+e] = (_Float16)xp[8+e];
  return a;
}

__device__ __forceinline__ v16h loadB_f16(const _Float16* __restrict__ p) {
  // 16 contiguous f16 (32B, 16B-aligned) -> two b128 loads, no conversion
  const v8h* vp = (const v8h*)p;
  v8h lo = vp[0], hi = vp[1];
  return __builtin_shufflevector(lo, hi, 0,1,2,3,4,5,6,7,8,9,10,11,12,13,14,15);
}

// ---------------------------------------------------------------------------
// Kernel 1: GX[s, m, n] = sum_k text[s,m,k] * Wall[s,n,k] + bWall[s,n]
// M=1200, K=512, N=256, batched over S. grid = S*(M/16) blocks of 128 threads
// (4 waves); wave w owns N cols [w*64, w*64+64) -> 4 wmma per K step, with
// register double-buffering of the next step's A/B fragments.
// ---------------------------------------------------------------------------
__global__ __launch_bounds__(128) void gemm_gx_wmma(
    const float* __restrict__ text, const _Float16* __restrict__ Wh,
    const float* __restrict__ bW, float* __restrict__ GX) {
  const int tid  = threadIdx.x;
  const int lane = tid & 31;
  const int wave = tid >> 5;
  const int blk  = blockIdx.x;
  const int s    = blk / (M_TL/16);
  const int mt   = blk % (M_TL/16);

  const float*    X = text + (size_t)s * M_TL * FT_;
  const _Float16* W = Wh   + (size_t)s * G4  * FT_;

  const int row   = mt*16 + (lane & 15);      // A row (M)
  const int n0    = wave * 64;                // this wave's N base
  const int halfA = (lane < 16) ? 0 : 16;     // A K-phase: contiguous k = 16h + s, same as B
  const int halfB = (lane < 16) ? 0 : 16;     // B K-phase: contiguous k = 16h + s, same as A

  const float* xbase = X + (size_t)row * FT_ + halfA;
  const _Float16* wbase[4];
#pragma unroll
  for (int j = 0; j < 4; ++j)
    wbase[j] = W + (size_t)(n0 + j*16 + (lane & 15)) * FT_ + halfB;

  v8f acc[4] = {};
  // prologue: fragments for k0 = 0
  v16h a = loadA_f32(xbase);
  v16h b[4];
#pragma unroll
  for (int j = 0; j < 4; ++j) b[j] = loadB_f16(wbase[j]);

  for (int k0 = 0; k0 < FT_; k0 += 32) {
    const int k1 = (k0 + 32 < FT_) ? (k0 + 32) : 0;   // wrap: keeps loads in-bounds
    __builtin_prefetch(xbase + k1 + 32, 0, 1);
    // fetch next step's fragments while wmma consumes the current ones
    v16h an = loadA_f32(xbase + k1);
    v16h bn[4];
#pragma unroll
    for (int j = 0; j < 4; ++j) bn[j] = loadB_f16(wbase[j] + k1);
#pragma unroll
    for (int j = 0; j < 4; ++j)
      acc[j] = __builtin_amdgcn_wmma_f32_16x16x32_f16(
          false, a, false, b[j], (short)0, acc[j], false, false);
    a = an;
#pragma unroll
    for (int j = 0; j < 4; ++j) b[j] = bn[j];
  }

  // store D + bias; C/D layout: VGPR v -> M = v + (lane<16?0:8), N = lane&15
  float* gx = GX + (size_t)s * M_TL * G4;
#pragma unroll
  for (int j = 0; j < 4; ++j) {
    const int ncol = n0 + j*16 + (lane & 15);
    const float bias = bW[s*G4 + ncol];
#pragma unroll
    for (int v = 0; v < 8; ++v) {
      const int m = mt*16 + v + ((lane < 16) ? 0 : 8);
      gx[(size_t)m * G4 + ncol] = acc[j][v] + bias;
    }
  }
}

// ---------------------------------------------------------------------------
// Kernel 2: TimeLSTM scan (T=40 steps) + temporal attention per (stock, day).
// One 256-thread workgroup per (s,d). Each thread owns one gate row of Uall
// (and one row of Wd for tid<64), hoisted into registers: weights are read
// from memory once per block instead of once per step (kills the 7.7 GB of
// repeated L2 traffic); h/c/outs live in LDS.
// ---------------------------------------------------------------------------
__global__ __launch_bounds__(256) void timelstm_attn(
    const float* __restrict__ GX, const float* __restrict__ Uall,
    const float* __restrict__ bU, const float* __restrict__ Wd,
    const float* __restrict__ bd, const float* __restrict__ taW,
    const float* __restrict__ tab, const float* __restrict__ ts,
    float* __restrict__ news) {
  __shared__ float h[H_], c[H_], cadj[H_], gs[G4], outs[T_*H_], av[H_], sc[T_], invs;
  const int tid = threadIdx.x;
  const int sd  = blockIdx.x;          // s*D + d
  const int s   = sd / D_;

  const float* U   = Uall + (size_t)s * G4 * H_;
  const float* wd  = Wd   + (size_t)s * H_ * H_;
  const float* gxp = GX   + (size_t)sd * T_ * G4;
  const float* tsp = ts   + (size_t)sd * T_;

  // hoist this thread's weight rows into registers
  float ur[H_];
#pragma unroll
  for (int j = 0; j < H_; ++j) ur[j] = U[(size_t)tid * H_ + j];
  const float bur = bU[s*G4 + tid];

  float wdr[H_];
  float bdr = 0.f;
  if (tid < H_) {
#pragma unroll
    for (int j = 0; j < H_; ++j) wdr[j] = wd[(size_t)tid * H_ + j];
    bdr = bd[s*H_ + tid];
  }

  if (tid < H_) { h[tid] = 0.f; c[tid] = 0.f; }
  __syncthreads();

  for (int t = 0; t < T_; ++t) {
    if (tid < H_) {                                   // c_s1 = tanh(Wd@c + bd)
      float a = bdr;
#pragma unroll
      for (int j = 0; j < H_; ++j) a += wdr[j] * c[j];
      float cs1 = tanhf(a);
      cadj[tid] = c[tid] - cs1 + cs1 * tsp[t];
    }
    __syncthreads();
    {                                                 // g = Uall@h + bU + gx
      float a = bur + gxp[t*G4 + tid];
#pragma unroll
      for (int j = 0; j < H_; ++j) a += ur[j] * h[j];
      gs[tid] = sigf(a);                              // all 4 gates sigmoid
    }
    __syncthreads();
    if (tid < H_) {
      float f  = gs[tid], ii = gs[H_+tid], o = gs[2*H_+tid], ct = gs[3*H_+tid];
      float c2 = f * cadj[tid] + ii * ct;
      float h2 = o * tanhf(c2);
      c[tid] = c2; h[tid] = h2; outs[t*H_ + tid] = h2;
    }
    __syncthreads();
  }

  // attention: scores = outs @ (taW@h + tab); softmax; news = a @ outs
  if (tid < H_) {
    float a = tab[s*H_ + tid];
    const float* wr = taW + (size_t)s * H_ * H_ + tid * H_;
    for (int j = 0; j < H_; ++j) a += wr[j] * h[j];
    av[tid] = a;
  }
  __syncthreads();
  if (tid < T_) {
    float a = 0.f;
    for (int j = 0; j < H_; ++j) a += outs[tid*H_ + j] * av[j];
    sc[tid] = a;
  }
  __syncthreads();
  if (tid == 0) {
    float mx = sc[0];
    for (int t = 1; t < T_; ++t) mx = fmaxf(mx, sc[t]);
    float sum = 0.f;
    for (int t = 0; t < T_; ++t) { sc[t] = __expf(sc[t] - mx); sum += sc[t]; }
    invs = 1.f / sum;
  }
  __syncthreads();
  if (tid < H_) {
    float a = 0.f;
    for (int t = 0; t < T_; ++t) a += sc[t] * outs[t*H_ + tid];
    news[(size_t)sd * H_ + tid] = a * invs;
  }
}

// ---------------------------------------------------------------------------
// Kernel 3/4: GRU scan (D=30 steps) + attention per stock; IN = input width.
// One 192-thread workgroup per stock; per-thread gate rows of Wih/Whh are
// register-resident across the scan.
// ---------------------------------------------------------------------------
template <int IN>
__global__ __launch_bounds__(192) void gru_attn(
    const float* __restrict__ xseq, const float* __restrict__ Wih,
    const float* __restrict__ Whh, const float* __restrict__ bih,
    const float* __restrict__ bhh, const float* __restrict__ aW,
    const float* __restrict__ ab, float* __restrict__ outv) {
  __shared__ float h[H_], gx[G3], gh[G3], outs[D_*H_], av[H_], sc[D_], xb[IN], invs;
  const int tid = threadIdx.x;
  const int s   = blockIdx.x;

  float wir[IN], whr[H_];
#pragma unroll
  for (int j = 0; j < IN; ++j) wir[j] = Wih[((size_t)s * G3 + tid) * IN + j];
#pragma unroll
  for (int j = 0; j < H_; ++j) whr[j] = Whh[((size_t)s * G3 + tid) * H_ + j];
  const float bir = bih[s*G3 + tid];
  const float bhr = bhh[s*G3 + tid];

  if (tid < H_) h[tid] = 0.f;
  __syncthreads();

  for (int d = 0; d < D_; ++d) {
    if (tid < IN) xb[tid] = xseq[((size_t)s * D_ + d) * IN + tid];
    __syncthreads();
    {
      float ax = bir;
#pragma unroll
      for (int j = 0; j < IN; ++j) ax += wir[j] * xb[j];
      gx[tid] = ax;
      float ah = bhr;
#pragma unroll
      for (int j = 0; j < H_; ++j) ah += whr[j] * h[j];
      gh[tid] = ah;
    }
    __syncthreads();
    if (tid < H_) {
      float r = sigf(gx[tid]        + gh[tid]);
      float z = sigf(gx[H_+tid]     + gh[H_+tid]);
      float n = tanhf(gx[2*H_+tid]  + r * gh[2*H_+tid]);
      float h2 = (1.f - z) * n + z * h[tid];
      h[tid] = h2; outs[d*H_ + tid] = h2;
    }
    __syncthreads();
  }

  if (tid < H_) {
    float a = ab[s*H_ + tid];
    const float* wr = aW + (size_t)s * H_ * H_ + tid * H_;
    for (int j = 0; j < H_; ++j) a += wr[j] * h[j];
    av[tid] = a;
  }
  __syncthreads();
  if (tid < D_) {
    float a = 0.f;
    for (int j = 0; j < H_; ++j) a += outs[tid*H_ + j] * av[j];
    sc[tid] = a;
  }
  __syncthreads();
  if (tid == 0) {
    float mx = sc[0];
    for (int d = 1; d < D_; ++d) mx = fmaxf(mx, sc[d]);
    float sum = 0.f;
    for (int d = 0; d < D_; ++d) { sc[d] = __expf(sc[d] - mx); sum += sc[d]; }
    invs = 1.f / sum;
  }
  __syncthreads();
  if (tid < H_) {
    float a = 0.f;
    for (int d = 0; d < D_; ++d) a += sc[d] * outs[d*H_ + tid];
    outv[s*H_ + tid] = a * invs;
  }
}

// ---------------------------------------------------------------------------
// Kernel 5: feature[s,k] = tanh( tvec[s] . biW[s,k] . xvec[s] + bib[s,k] )
// Memory-bound stream of the 105 MB (S,H,H,H) tensor; one block per stock.
// ---------------------------------------------------------------------------
__global__ __launch_bounds__(256) void bilinear_k(
    const float* __restrict__ tvec, const float* __restrict__ xvec,
    const float* __restrict__ biW, const float* __restrict__ bib,
    float* __restrict__ feat) {
  __shared__ float tv[H_], xv[H_], part[256];
  const int tid = threadIdx.x, s = blockIdx.x;
  if (tid < H_) { tv[tid] = tvec[s*H_+tid]; xv[tid] = xvec[s*H_+tid]; }
  __syncthreads();
  const int k  = tid >> 2;     // output row 0..63
  const int iq = tid & 3;      // quarter of i-range
  const float* Wb = biW + ((size_t)s * H_ + k) * H_ * H_;
  float acc = 0.f;
  for (int i = iq*16; i < iq*16 + 16; ++i) {
    const float* row = Wb + i * H_;
    float dot = 0.f;
#pragma unroll 8
    for (int j = 0; j < H_; ++j) dot += row[j] * xv[j];
    acc += tv[i] * dot;
  }
  part[tid] = acc;
  __syncthreads();
  if (iq == 0) {
    float v = part[tid] + part[tid+1] + part[tid+2] + part[tid+3];
    feat[s*H_ + k] = tanhf(v + bib[s*H_ + k]);
  }
}

// ---------------------------------------------------------------------------
// Kernel 6: out1 = tanh(feature @ bl_W[-1]^T + bl_b[-1]); Q/K projections.
// Also zeroes the loss accumulator (d_out[0]) before kernel 7 runs.
// ---------------------------------------------------------------------------
__global__ __launch_bounds__(64) void proj_qk(
    const float* __restrict__ feat, const float* __restrict__ blW,
    const float* __restrict__ blb, const float* __restrict__ Wq,
    const float* __restrict__ bq, const float* __restrict__ Wk,
    const float* __restrict__ bk, float* __restrict__ out1,
    float* __restrict__ Q, float* __restrict__ K, float* __restrict__ dout) {
  __shared__ float f[H_];
  const int tid = threadIdx.x, s = blockIdx.x;
  f[tid] = feat[s*H_ + tid];
  __syncthreads();
  float aq = bq[tid], ak = bk[tid];
  for (int j = 0; j < H_; ++j) {
    aq += Wq[tid*H_ + j] * f[j];
    ak += Wk[tid*H_ + j] * f[j];
  }
  Q[s*H_ + tid] = aq;
  K[s*H_ + tid] = ak;
  if (tid < 2) {
    const float* w = blW + ((size_t)(S_-1)*2 + tid) * H_;
    float a = blb[(S_-1)*2 + tid];
    for (int j = 0; j < H_; ++j) a += w[j] * f[j];
    out1[s*2 + tid] = tanhf(a);
  }
  if (s == 0 && tid == 0) dout[0] = 0.f;
}

// ---------------------------------------------------------------------------
// Kernel 7: per-stock MHA over all S stocks + elu + softmax + loss.
// One 128-thread block per query stock; 32 lanes per head.
// ---------------------------------------------------------------------------
__global__ __launch_bounds__(128) void mha_loss(
    const float* __restrict__ feat, const float* __restrict__ Q,
    const float* __restrict__ K, const float* __restrict__ out1,
    const float* __restrict__ fcW, const float* __restrict__ fcb,
    const int* __restrict__ label, float* __restrict__ dout) {
  __shared__ float q[H_], sc[NH_][S_], mh[H_], xo[2], o1[2];
  const int tid = threadIdx.x, s = blockIdx.x;
  const int hh = tid >> 5, lane = tid & 31;
  if (tid < H_) q[tid] = Q[s*H_ + tid];
  if (tid < 2)  o1[tid] = out1[s*2 + tid];
  __syncthreads();
  for (int t = lane; t < S_; t += 32) {
    float a = 0.f;
#pragma unroll
    for (int d = 0; d < DH_; ++d) a += q[hh*DH_ + d] * K[t*H_ + hh*DH_ + d];
    sc[hh][t] = a * 0.25f;                 // 1/sqrt(DH)
  }
  __syncthreads();
  if (lane == 0) {
    float mx = -1e30f;
    for (int t = 0; t < S_; ++t) mx = fmaxf(mx, sc[hh][t]);
    float sum = 0.f;
    for (int t = 0; t < S_; ++t) { sc[hh][t] = __expf(sc[hh][t] - mx); sum += sc[hh][t]; }
    float inv = 1.f / sum;
    for (int t = 0; t < S_; ++t) sc[hh][t] *= inv;
  }
  __syncthreads();
  if (lane < DH_) {
    float a = 0.f;
    for (int t = 0; t < S_; ++t) a += sc[hh][t] * feat[t*H_ + hh*DH_ + lane];
    mh[hh*DH_ + lane] = a;
  }
  __syncthreads();
  if (tid < 2) {
    float a = fcb[tid];
    for (int j = 0; j < H_; ++j) a += fcW[tid*H_ + j] * mh[j];
    float e = (a > 0.f) ? a : (__expf(a) - 1.f);   // elu
    xo[tid] = e + o1[tid];
  }
  __syncthreads();
  if (tid == 0) {
    float m  = fmaxf(xo[0], xo[1]);
    float e0 = __expf(xo[0] - m), e1 = __expf(xo[1] - m);
    float inv = 1.f / (e0 + e1);
    float p0 = e0 * inv, p1 = e1 * inv;
    dout[1 + 2*s]     = p0;                // output = softmax(x + out1)
    dout[1 + 2*s + 1] = p1;
    // logp = log_softmax(output)
    float mm = fmaxf(p0, p1);
    float lse = mm + logf(__expf(p0 - mm) + __expf(p1 - mm));
    float l0 = p0 - lse, l1 = p1 - lse;
    int lbl = label[s];
    atomicAdd(dout, -((lbl == 0) ? l0 : l1) / (float)S_);
  }
}

// ---------------------------------------------------------------------------
extern "C" void kernel_launch(void* const* d_in, const int* in_sizes, int n_in,
                              void* d_out, int out_size, void* d_ws, size_t ws_size,
                              hipStream_t stream) {
  const float* pg_Wih  = (const float*)d_in[0];
  const float* pg_Whh  = (const float*)d_in[1];
  const float* pg_bih  = (const float*)d_in[2];
  const float* pg_bhh  = (const float*)d_in[3];
  const float* pa_W    = (const float*)d_in[4];
  const float* pa_b    = (const float*)d_in[5];
  const float* tl_Wall = (const float*)d_in[6];
  const float* tl_bWall= (const float*)d_in[7];
  const float* tl_Uall = (const float*)d_in[8];
  const float* tl_bUall= (const float*)d_in[9];
  const float* tl_Wd   = (const float*)d_in[10];
  const float* tl_bd   = (const float*)d_in[11];
  const float* ta_W    = (const float*)d_in[12];
  const float* ta_b    = (const float*)d_in[13];
  const float* sg_Wih  = (const float*)d_in[14];
  const float* sg_Whh  = (const float*)d_in[15];
  const float* sg_bih  = (const float*)d_in[16];
  const float* sg_bhh  = (const float*)d_in[17];
  const float* sa_W    = (const float*)d_in[18];
  const float* sa_b    = (const float*)d_in[19];
  const float* bi_W    = (const float*)d_in[20];
  const float* bi_b    = (const float*)d_in[21];
  const float* bl_W    = (const float*)d_in[22];
  const float* bl_b    = (const float*)d_in[23];
  const float* mha_Wq  = (const float*)d_in[24];
  const float* mha_bq  = (const float*)d_in[25];
  const float* mha_Wk  = (const float*)d_in[26];
  const float* mha_bk  = (const float*)d_in[27];
  const float* fc_W    = (const float*)d_in[28];
  const float* fc_b    = (const float*)d_in[29];
  const float* text    = (const float*)d_in[30];
  const float* price   = (const float*)d_in[31];
  const float* tstamps = (const float*)d_in[32];
  const int*   label   = (const int*)  d_in[33];
  (void)in_sizes; (void)n_in; (void)out_size; (void)ws_size;

  float* ws = (float*)d_ws;
  size_t off = 0;
  float* GX   = ws + off; off += (size_t)S_ * M_TL * G4;      // 30.72M floats
  _Float16* WallH = (_Float16*)(ws + off);
  off += ((size_t)S_ * G4 * FT_) / 2;                          // 13.1M halves
  float* news = ws + off; off += (size_t)S_ * D_ * H_;
  float* xvec = ws + off; off += (size_t)S_ * H_;
  float* tvec = ws + off; off += (size_t)S_ * H_;
  float* feat = ws + off; off += (size_t)S_ * H_;
  float* out1 = ws + off; off += (size_t)S_ * 2;
  float* Qb   = ws + off; off += (size_t)S_ * H_;
  float* Kb   = ws + off; off += (size_t)S_ * H_;

  float* out = (float*)d_out;

  const int n_wall = S_ * G4 * FT_;                            // 13,107,200
  cvt_wall_f16<<<n_wall / (256*4), 256, 0, stream>>>(tl_Wall, WallH);
  gemm_gx_wmma<<<S_ * (M_TL/16), 128, 0, stream>>>(text, WallH, tl_bWall, GX);
  timelstm_attn<<<S_ * D_, 256, 0, stream>>>(GX, tl_Uall, tl_bUall, tl_Wd, tl_bd,
                                             ta_W, ta_b, tstamps, news);
  gru_attn<FP_><<<S_, 192, 0, stream>>>(price, pg_Wih, pg_Whh, pg_bih, pg_bhh,
                                        pa_W, pa_b, xvec);
  gru_attn<H_><<<S_, 192, 0, stream>>>(news, sg_Wih, sg_Whh, sg_bih, sg_bhh,
                                       sa_W, sa_b, tvec);
  bilinear_k<<<S_, 256, 0, stream>>>(tvec, xvec, bi_W, bi_b, feat);
  proj_qk<<<S_, 64, 0, stream>>>(feat, bl_W, bl_b, mha_Wq, mha_bq, mha_Wk, mha_bk,
                                 out1, Qb, Kb, out);
  mha_loss<<<S_, 128, 0, stream>>>(feat, Qb, Kb, out1, fc_W, fc_b, label, out);
}
